// Adjustor_36386962931813
// MI455X (gfx1250) — matrix operand order repaired and checked by reading, not compiled, not run on hardware
//
#include <hip/hip_runtime.h>
#include <hip/hip_bf16.h>

// ---------------------------------------------------------------------------
// MI455X / gfx1250: bf16 WMMA flash-attention + fused MLP.
// - All B (K-major) operands pre-swizzled into fragment-ready order: every
//   WMMA operand load is b128-vectorized.
// - Attention K/V tiles staged with double-buffered async global->LDS copies
//   (GLOBAL_LOAD_ASYNC_TO_LDS_B128 / ASYNCcnt), overlapped with WMMA compute.
// Workspace requirement: ~3.5 MB.
// ---------------------------------------------------------------------------

typedef __attribute__((ext_vector_type(16))) __bf16 v16bf;
typedef __attribute__((ext_vector_type(8)))  __bf16 v8bf;
typedef __attribute__((ext_vector_type(8)))  float  v8f;

#define EMBED   384
#define DHEAD   64
#define TROWS   4096
#define SROWS   8192
#define H1DIM   128
#define H2DIM   64
#define NCLS    5
#define BS      64    // attention S-tile (double-buffered)

static __device__ __forceinline__ v8f wmma_bf16(v16bf a, v16bf b, v8f c) {
  return __builtin_amdgcn_wmma_f32_16x16x32_bf16(false, a, false, b, (short)0, c,
                                                 false, false);
}

// ---- async global->LDS (CDNA5, tracked by ASYNCcnt) -----------------------
static __device__ __forceinline__ void async_ld16(unsigned lds_off, const void* gptr) {
  asm volatile("global_load_async_to_lds_b128 %0, %1, off"
               :: "v"(lds_off), "v"(gptr) : "memory");
}
static __device__ __forceinline__ void wait_async0() {
  asm volatile("s_wait_asynccnt 0x0" ::: "memory");
}
// stage an 8 KB tile (512 x 16B) with 128 threads, 4 async b128 per lane
static __device__ __forceinline__ void stage_tile_async(__bf16* lbuf,
                                                        const __bf16* gsrc, int tid) {
  unsigned base = (unsigned)(uintptr_t)lbuf;  // low 32 bits = LDS offset
#pragma unroll
  for (int i = 0; i < 4; ++i)
    async_ld16(base + (unsigned)((tid + 128 * i) * 16), gsrc + (tid + 128 * i) * 8);
}

// ---- fragment loaders -----------------------------------------------------
// A 16x32 bf16 from row-major [16][>=32]: one contiguous 16-elt run (k = 16h + s) -> 2x b128.
static __device__ __forceinline__ v16bf load_a_bf16(const __bf16* src, int ld) {
  const int lane = threadIdx.x & 31;
  const __bf16* p = src + (lane & 15) * ld + ((lane >> 4) << 4);
  v8bf lo = *(const v8bf*)(p);
  v8bf hi = *(const v8bf*)(p + 8);
  return __builtin_shufflevector(lo, hi, 0, 1, 2, 3, 4, 5, 6, 7,
                                 8, 9, 10, 11, 12, 13, 14, 15);
}

// A 16x32 from fp32 row-major: 2x v8f vector loads + pk converts.
static __device__ __forceinline__ v16bf load_a_f32(const float* src, int ld) {
  const int lane = threadIdx.x & 31;
  const float* p = src + (lane & 15) * ld + ((lane >> 4) << 4);
  v8f lo = *(const v8f*)(p);
  v8f hi = *(const v8f*)(p + 8);
  v16bf f;
#pragma unroll
  for (int j = 0; j < 8; ++j) {
    f[j]     = (__bf16)lo[j];
    f[8 + j] = (__bf16)hi[j];
  }
  return f;
}

// B 32x16 where source is N-major [N][K] (B^T row-major): 16 contiguous bf16
// per lane -> one 32B vector load. Used for Q @ K^T.
static __device__ __forceinline__ v16bf load_b_nt(const __bf16* src, int ld) {
  const int lane = threadIdx.x & 31;
  const __bf16* p = src + (lane & 15) * ld + ((lane >> 4) ? 16 : 0);
  return *(const v16bf*)p;
}

// B 32x16 from fragment-ready (swizzled) storage.
static __device__ __forceinline__ v16bf load_b_sw(const __bf16* base, int frag) {
  const int lane = threadIdx.x & 31;
  return *(const v16bf*)(base + ((size_t)frag * 32 + lane) * 16);
}

// Reductions across the 16-lane half-group.
static __device__ __forceinline__ float rmax16(float v) {
  v = fmaxf(v, __shfl_xor(v, 1, 32));
  v = fmaxf(v, __shfl_xor(v, 2, 32));
  v = fmaxf(v, __shfl_xor(v, 4, 32));
  v = fmaxf(v, __shfl_xor(v, 8, 32));
  return v;
}
static __device__ __forceinline__ float rsum16(float v) {
  v += __shfl_xor(v, 1, 32);
  v += __shfl_xor(v, 2, 32);
  v += __shfl_xor(v, 4, 32);
  v += __shfl_xor(v, 8, 32);
  return v;
}

// ---------------------------------------------------------------------------
// Kernel 0: fp32 -> bf16 conversion with B-fragment swizzle.
// ---------------------------------------------------------------------------
__global__ void swz_w_bf16(const float* __restrict__ src, __bf16* __restrict__ dst,
                           int N, int n) {
  int idx = blockIdx.x * 256 + threadIdx.x;
  if (idx >= n) return;
  int e = idx & 15;
  int L = (idx >> 4) & 31;
  int t = idx >> 9;
  int ntiles = N >> 4;
  int nt = t % ntiles;
  int kt = t / ntiles;
  int row = kt * 32 + (L >> 4) * 16 + e;
  int col = nt * 16 + (L & 15);
  dst[idx] = (__bf16)src[(size_t)row * N + col];
}

// ---------------------------------------------------------------------------
// Kernel 1: projection  Y = (X[rows,384] @ W + b) * scale
// vswz=0 -> Y row-major bf16 [rows,64]; vswz=1 -> Y fragment-swizzled (V).
// ---------------------------------------------------------------------------
__global__ __launch_bounds__(128) void proj_kernel(
    const float* __restrict__ X, const __bf16* __restrict__ Wsw,
    const float* __restrict__ bias, __bf16* __restrict__ Y, float scale,
    int vswz) {
  __shared__ __align__(32) __bf16 Wl[EMBED * DHEAD];  // 48 KB, swizzled order

  const int tid = threadIdx.x;
  {
    const uint4* g = (const uint4*)Wsw;
    uint4* l = (uint4*)Wl;
#pragma unroll
    for (int i = 0; i < 24; ++i) l[tid + 128 * i] = g[tid + 128 * i];
  }
  __syncthreads();

  const int wave = tid >> 5;
  const int lane = tid & 31;
  const int rowbase = blockIdx.x * 64 + wave * 16;

  v8f acc[4];
#pragma unroll
  for (int nt = 0; nt < 4; ++nt) acc[nt] = v8f{0.f, 0.f, 0.f, 0.f, 0.f, 0.f, 0.f, 0.f};

#pragma unroll 4
  for (int ks = 0; ks < EMBED / 32; ++ks) {
    v16bf a = load_a_f32(X + (size_t)rowbase * EMBED + ks * 32, EMBED);
#pragma unroll
    for (int nt = 0; nt < 4; ++nt) {
      v16bf b = load_b_sw(Wl, ks * 4 + nt);
      acc[nt] = wmma_bf16(a, b, acc[nt]);
    }
  }

  const int col16 = lane & 15;
  const int rb = (lane >> 4) * 8;
#pragma unroll
  for (int nt = 0; nt < 4; ++nt) {
    int c = nt * 16 + col16;
    float bb = bias[c];
#pragma unroll
    for (int i = 0; i < 8; ++i) {
      float v = (acc[nt][i] + bb) * scale;
      int s = rowbase + rb + i;
      if (!vswz) {
        Y[(size_t)s * DHEAD + c] = (__bf16)v;
      } else {
        size_t off = ((size_t)((s >> 5) * 4 + nt) * 32 + ((s >> 4) & 1) * 16 + col16) * 16
                     + (s & 15);
        Y[off] = (__bf16)v;
      }
    }
  }
}

// ---------------------------------------------------------------------------
// Kernel 2: flash attention, double-buffered async K/V staging.
// 4 waves, 64 T rows per block, S tiles of 64. Q pre-scaled by 1/8.
// ---------------------------------------------------------------------------
__global__ __launch_bounds__(128) void attn_kernel(
    const __bf16* __restrict__ Qbf, const __bf16* __restrict__ Kbf,
    const __bf16* __restrict__ Vsw, __bf16* __restrict__ Abf) {
  __shared__ __align__(32) __bf16 Kt[2][BS * DHEAD];    // 2 x 8 KB (row-major)
  __shared__ __align__(32) __bf16 Vt[2][BS * DHEAD];    // 2 x 8 KB (fragment order)
  __shared__ __align__(32) __bf16 Pb[4 * 16 * BS];      // 8 KB (row-major P)

  const int tid = threadIdx.x;
  const int wave = tid >> 5;
  const int lane = tid & 31;
  const int rowbase = blockIdx.x * 64 + wave * 16;

  v16bf qa0 = load_a_bf16(Qbf + (size_t)rowbase * DHEAD, DHEAD);
  v16bf qa1 = load_a_bf16(Qbf + (size_t)rowbase * DHEAD + 32, DHEAD);

  v8f o[4];
#pragma unroll
  for (int nt = 0; nt < 4; ++nt) o[nt] = v8f{0.f, 0.f, 0.f, 0.f, 0.f, 0.f, 0.f, 0.f};
  float m[8], l[8];
#pragma unroll
  for (int i = 0; i < 8; ++i) { m[i] = -1e30f; l[i] = 0.f; }

  __bf16* pb = Pb + wave * 16 * BS;
  const int col16 = lane & 15;
  const int rb = (lane >> 4) * 8;
  const int NT = SROWS / BS;

  // prologue: async-stage tile 0 into buffer 0
  stage_tile_async(Kt[0], Kbf, tid);
  stage_tile_async(Vt[0], Vsw, tid);
  wait_async0();
  __syncthreads();

  for (int st = 0; st < NT; ++st) {
    const int cur = st & 1;
    if (st + 1 < NT) {  // overlap: issue next tile's async copies now
      stage_tile_async(Kt[cur ^ 1], Kbf + (size_t)(st + 1) * BS * DHEAD, tid);
      stage_tile_async(Vt[cur ^ 1], Vsw + (size_t)(st + 1) * BS * DHEAD, tid);
    }

    // scores S = Q @ K^T : 4 tiles of 16x16
    v8f sa[4];
#pragma unroll
    for (int nt = 0; nt < 4; ++nt) sa[nt] = v8f{0.f, 0.f, 0.f, 0.f, 0.f, 0.f, 0.f, 0.f};
#pragma unroll
    for (int nt = 0; nt < 4; ++nt) {
      v16bf b0 = load_b_nt(Kt[cur] + (nt * 16) * DHEAD + 0, DHEAD);
      sa[nt] = wmma_bf16(qa0, b0, sa[nt]);
      v16bf b1 = load_b_nt(Kt[cur] + (nt * 16) * DHEAD + 32, DHEAD);
      sa[nt] = wmma_bf16(qa1, b1, sa[nt]);
    }

    // online softmax
#pragma unroll
    for (int i = 0; i < 8; ++i) {
      float tm = fmaxf(fmaxf(sa[0][i], sa[1][i]), fmaxf(sa[2][i], sa[3][i]));
      tm = rmax16(tm);
      float nm = fmaxf(m[i], tm);
      float corr = __expf(m[i] - nm);
      m[i] = nm;
      float ts = 0.f;
#pragma unroll
      for (int nt = 0; nt < 4; ++nt) {
        float p = __expf(sa[nt][i] - nm);
        sa[nt][i] = p;
        ts += p;
      }
      ts = rsum16(ts);
      l[i] = l[i] * corr + ts;
#pragma unroll
      for (int nt = 0; nt < 4; ++nt) o[nt][i] *= corr;
    }

    // stage P (bf16) row-major [16][BS], wave-private region (compiler
    // inserts the same-wave s_wait_dscnt for the RAW below)
#pragma unroll
    for (int nt = 0; nt < 4; ++nt) {
      int c = nt * 16 + col16;
#pragma unroll
      for (int i = 0; i < 8; ++i) pb[(rb + i) * BS + c] = (__bf16)sa[nt][i];
    }

    // O += P @ V  (K-dim = BS)
#pragma unroll
    for (int kh = 0; kh < BS / 32; ++kh) {
      v16bf a = load_a_bf16(pb + kh * 32, BS);
#pragma unroll
      for (int nt = 0; nt < 4; ++nt) {
        v16bf b = load_b_sw(Vt[cur], kh * 4 + nt);
        o[nt] = wmma_bf16(a, b, o[nt]);
      }
    }

    // drain this iteration's async copies, then block-wide barrier before
    // anyone touches the freshly filled buffer (or re-fills this one)
    wait_async0();
    __syncthreads();
  }

  // normalize + write attended (bf16 row-major)
#pragma unroll
  for (int i = 0; i < 8; ++i) {
    float inv = 1.0f / l[i];
#pragma unroll
    for (int nt = 0; nt < 4; ++nt) {
      int c = nt * 16 + col16;
      Abf[(size_t)(rowbase + rb + i) * DHEAD + c] = (__bf16)(o[nt][i] * inv);
    }
  }
}

// ---------------------------------------------------------------------------
// Kernel 3: fused MLP. features = [attended(64) | tag(384) | flag(1)]
// ---------------------------------------------------------------------------
__global__ __launch_bounds__(256) void mlp_kernel(
    const __bf16* __restrict__ Abf, const float* __restrict__ tag,
    const float* __restrict__ flags, const __bf16* __restrict__ W1sw,
    const float* __restrict__ W1f, const float* __restrict__ b1,
    const __bf16* __restrict__ W2sw, const float* __restrict__ b2,
    const float* __restrict__ W3, const float* __restrict__ b3,
    float* __restrict__ out) {
  __shared__ __align__(32) __bf16 H1[8 * 16 * H1DIM];  // 32 KB
  __shared__ __align__(32) __bf16 H2[8 * 16 * H2DIM];  // 16 KB

  const int tid = threadIdx.x;
  const int wave = tid >> 5;
  const int lane = tid & 31;
  const int rowbase = blockIdx.x * 128 + wave * 16;
  const int col16 = lane & 15;
  const int rb = (lane >> 4) * 8;

  // ---- layer 1: [16 x 448] @ [448 x 128] ----
  v8f acc1[8];
#pragma unroll
  for (int nt = 0; nt < 8; ++nt) acc1[nt] = v8f{0.f, 0.f, 0.f, 0.f, 0.f, 0.f, 0.f, 0.f};

#pragma unroll
  for (int kh = 0; kh < 2; ++kh) {  // attended part, K 0..63
    v16bf a = load_a_bf16(Abf + (size_t)rowbase * DHEAD + kh * 32, DHEAD);
#pragma unroll
    for (int nt = 0; nt < 8; ++nt)
      acc1[nt] = wmma_bf16(a, load_b_sw(W1sw, kh * 8 + nt), acc1[nt]);
  }
  for (int ks = 0; ks < EMBED / 32; ++ks) {  // embedding part, K 64..447
    v16bf a = load_a_f32(tag + (size_t)rowbase * EMBED + ks * 32, EMBED);
#pragma unroll
    for (int nt = 0; nt < 8; ++nt)
      acc1[nt] = wmma_bf16(a, load_b_sw(W1sw, (2 + ks) * 8 + nt), acc1[nt]);
  }

  float flg[8];
#pragma unroll
  for (int i = 0; i < 8; ++i) flg[i] = flags[rowbase + rb + i];

  __bf16* h1 = H1 + wave * 16 * H1DIM;
#pragma unroll
  for (int nt = 0; nt < 8; ++nt) {
    int c = nt * 16 + col16;
    float w448 = W1f[448 * H1DIM + c];  // rank-1 flag column
    float bb = b1[c];
#pragma unroll
    for (int i = 0; i < 8; ++i) {
      float v = acc1[nt][i] + flg[i] * w448 + bb;
      h1[(rb + i) * H1DIM + c] = (__bf16)fmaxf(v, 0.f);
    }
  }
  __syncthreads();

  // ---- layer 2: [16 x 128] @ [128 x 64] ----
  v8f acc2[4];
#pragma unroll
  for (int nt = 0; nt < 4; ++nt) acc2[nt] = v8f{0.f, 0.f, 0.f, 0.f, 0.f, 0.f, 0.f, 0.f};
#pragma unroll
  for (int ks = 0; ks < 4; ++ks) {
    v16bf a = load_a_bf16(h1 + ks * 32, H1DIM);
#pragma unroll
    for (int nt = 0; nt < 4; ++nt)
      acc2[nt] = wmma_bf16(a, load_b_sw(W2sw, ks * 4 + nt), acc2[nt]);
  }
  __bf16* h2 = H2 + wave * 16 * H2DIM;
#pragma unroll
  for (int nt = 0; nt < 4; ++nt) {
    int c = nt * 16 + col16;
    float bb = b2[c];
#pragma unroll
    for (int i = 0; i < 8; ++i)
      h2[(rb + i) * H2DIM + c] = (__bf16)fmaxf(acc2[nt][i] + bb, 0.f);
  }
  __syncthreads();

  // ---- layer 3: [16 x 64] @ [64 x 5] (tiny; scalar) ----
  for (int t = lane; t < 16 * NCLS; t += 32) {
    int r = t / NCLS, c = t % NCLS;
    float s = b3[c];
#pragma unroll 8
    for (int j = 0; j < H2DIM; ++j) s += (float)h2[r * H2DIM + j] * W3[j * NCLS + c];
    out[(size_t)(rowbase + r) * NCLS + c] = s;
  }
}

// ---------------------------------------------------------------------------
// Launch
// ---------------------------------------------------------------------------
extern "C" void kernel_launch(void* const* d_in, const int* in_sizes, int n_in,
                              void* d_out, int out_size, void* d_ws, size_t ws_size,
                              hipStream_t stream) {
  (void)in_sizes; (void)n_in; (void)out_size; (void)ws_size;
  const float* prompt = (const float*)d_in[0];
  const float* tag    = (const float*)d_in[1];
  const float* flags  = (const float*)d_in[2];
  const float* Wq = (const float*)d_in[3];  const float* bq = (const float*)d_in[4];
  const float* Wk = (const float*)d_in[5];  const float* bk = (const float*)d_in[6];
  const float* Wv = (const float*)d_in[7];  const float* bv = (const float*)d_in[8];
  const float* W1 = (const float*)d_in[9];  const float* b1 = (const float*)d_in[10];
  const float* W2 = (const float*)d_in[11]; const float* b2 = (const float*)d_in[12];
  const float* W3 = (const float*)d_in[13]; const float* b3 = (const float*)d_in[14];
  float* out = (float*)d_out;

  char* ws = (char*)d_ws;
  __bf16* Qbf  = (__bf16*)(ws + 0);          // 512 KB
  __bf16* Kbf  = (__bf16*)(ws + 524288);     // 1 MB  (row-major)
  __bf16* Vsw  = (__bf16*)(ws + 1572864);    // 1 MB  (fragment order)
  __bf16* Abf  = (__bf16*)(ws + 2621440);    // 512 KB
  __bf16* Wqsw = (__bf16*)(ws + 3145728);    // 48 KB
  __bf16* Wksw = (__bf16*)(ws + 3194880);    // 48 KB
  __bf16* Wvsw = (__bf16*)(ws + 3244032);    // 48 KB
  __bf16* W1sw = (__bf16*)(ws + 3293184);    // 448*128*2 = 112 KB
  __bf16* W2sw = (__bf16*)(ws + 3407872);    // 16 KB

  auto swz = [&](const float* s, __bf16* d, int N, int n) {
    swz_w_bf16<<<(n + 255) / 256, 256, 0, stream>>>(s, d, N, n);
  };
  swz(Wq, Wqsw, DHEAD, EMBED * DHEAD);
  swz(Wk, Wksw, DHEAD, EMBED * DHEAD);
  swz(Wv, Wvsw, DHEAD, EMBED * DHEAD);
  swz(W1, W1sw, H1DIM, 448 * H1DIM);
  swz(W2, W2sw, H2DIM, H1DIM * H2DIM);

  proj_kernel<<<TROWS / 64, 128, 0, stream>>>(tag,    Wqsw, bq, Qbf, 0.125f, 0);
  proj_kernel<<<SROWS / 64, 128, 0, stream>>>(prompt, Wksw, bk, Kbf, 1.0f,   0);
  proj_kernel<<<SROWS / 64, 128, 0, stream>>>(prompt, Wvsw, bv, Vsw, 1.0f,   1);

  attn_kernel<<<TROWS / 64, 128, 0, stream>>>(Qbf, Kbf, Vsw, Abf);

  mlp_kernel<<<TROWS / 128, 256, 0, stream>>>(Abf, tag, flags, W1sw, W1, b1,
                                              W2sw, b2, W3, b3, out);
}
